// ProteinFeatureExtractor_65841848648305
// MI455X (gfx1250) — matrix operand order repaired and checked by reading, not compiled, not run on hardware
//
#include <hip/hip_runtime.h>
#include <hip/hip_bf16.h>
#include <cstdint>

// ---------- CDNA5 WMMA types ----------
typedef __bf16  v16bf __attribute__((ext_vector_type(16)));
typedef float   v8f   __attribute__((ext_vector_type(8)));
typedef int     v8i   __attribute__((ext_vector_type(8)));

// ---------- problem constants ----------
#define NN      10000
#define EE      160000
#define EA      170000        // E + N self loops
#define K1P     512           // 481 padded to 512
#define DMID    2048
#define H1H     8
#define O1O     256
#define H2H     3
#define O2O     481
#define DPOOL   1443
#define LD2     1472          // 1443 padded to 64-multiple
#define K2K     2048

__device__ __forceinline__ unsigned short f2bf(float f) {
    unsigned u = __builtin_bit_cast(unsigned, f);
    unsigned r = u + 0x7FFFu + ((u >> 16) & 1u);   // round-to-nearest-even
    return (unsigned short)(r >> 16);
}
__device__ __forceinline__ unsigned enc_f(float f) {
    unsigned u = __builtin_bit_cast(unsigned, f);
    return (u & 0x80000000u) ? ~u : (u | 0x80000000u);
}
__device__ __forceinline__ float dec_f(unsigned e) {
    unsigned u = (e & 0x80000000u) ? (e & 0x7FFFFFFFu) : ~e;
    return __builtin_bit_cast(float, u);
}

// ---------- prep kernels ----------
__global__ void k_build_x1(const float* __restrict__ tok, const float* __restrict__ pos,
                           unsigned short* __restrict__ X) {
    long long total = (long long)NN * K1P;
    for (long long i = blockIdx.x * (long long)blockDim.x + threadIdx.x; i < total;
         i += (long long)gridDim.x * blockDim.x) {
        int r = (int)(i >> 9), c = (int)(i & 511);
        float v = 0.f;
        if (c < 478)      v = tok[(long long)r * 478 + c];
        else if (c < 481) v = pos[(long long)r * 3 + (c - 478)];
        X[i] = f2bf(v);
    }
}

// Wt[nc,k] = W[k,nc] (zero pad); W is [K0,N0] row-major, Wt is [Np,Kp]
__global__ void k_prep_wt(const float* __restrict__ W, unsigned short* __restrict__ Wt,
                          int N0, int K0, int Np, int Kp) {
    long long total = (long long)Np * Kp;
    for (long long i = blockIdx.x * (long long)blockDim.x + threadIdx.x; i < total;
         i += (long long)gridDim.x * blockDim.x) {
        int nc = (int)(i / Kp), k = (int)(i % Kp);
        float v = (nc < N0 && k < K0) ? W[(long long)k * N0 + nc] : 0.f;
        Wt[i] = f2bf(v);
    }
}

// ---------- bf16 WMMA GEMM: C[M,Ncol] = A[M,K] * Bt[Ncol,K]^T ----------
// One wave per 16x64 output tile (4 accumulators sharing one A fragment),
// register double-buffered: fragments for step s+1 are in flight while the
// four WMMAs of step s execute.
__device__ __forceinline__ v8i load_a_frag(const uint32_t* __restrict__ ap) {
    v8i ai;
    ai[0] = ap[0];  ai[1] = ap[1];  ai[2] = ap[2];  ai[3] = ap[3];
    ai[4] = ap[4];  ai[5] = ap[5];  ai[6] = ap[6];  ai[7] = ap[7];
    return ai;
}
__device__ __forceinline__ v8i load_b_frag(const uint32_t* __restrict__ bp) {
    v8i bi;
    bi[0] = bp[0]; bi[1] = bp[1]; bi[2] = bp[2]; bi[3] = bp[3];
    bi[4] = bp[4]; bi[5] = bp[5]; bi[6] = bp[6]; bi[7] = bp[7];
    return bi;
}
__device__ __forceinline__ v8f wmma_bf16(v8i a, v8i b, v8f c) {
    return __builtin_amdgcn_wmma_f32_16x16x32_bf16(
        false, __builtin_bit_cast(v16bf, a),
        false, __builtin_bit_cast(v16bf, b),
        (short)0, c, false, false);
}

__global__ void k_gemm_bf16(const unsigned short* __restrict__ A,
                            const unsigned short* __restrict__ Bt,
                            float* __restrict__ C, int M, int Ncol, int K) {
    int wid  = (int)((blockIdx.x * (long long)blockDim.x + threadIdx.x) >> 5);
    int lane = threadIdx.x & 31;
    int tn = Ncol >> 6, tm = M >> 4;
    if (wid >= tm * tn) return;
    int ti = wid / tn, tj = wid % tn;
    int m0 = ti << 4, n0 = tj << 6;

    const uint32_t* A32 = (const uint32_t*)A;
    const uint32_t* B32 = (const uint32_t*)Bt;

    int arow = m0 + (lane & 15);
    int akb  = (lane >> 4) << 4;    // K-half base for A: 0 or 16
    int bn   = lane & 15;           // B column within tile
    int bkh  = (lane >> 4) << 4;    // K-half base for B: 0 or 16

    const uint32_t* ap  = A32 + (((long long)arow * K + akb) >> 1);
    const uint32_t* bp0 = B32 + ((((long long)(n0 +  0 + bn)) * K + bkh) >> 1);
    const uint32_t* bp1 = B32 + ((((long long)(n0 + 16 + bn)) * K + bkh) >> 1);
    const uint32_t* bp2 = B32 + ((((long long)(n0 + 32 + bn)) * K + bkh) >> 1);
    const uint32_t* bp3 = B32 + ((((long long)(n0 + 48 + bn)) * K + bkh) >> 1);

    v8f zero = {0.f,0.f,0.f,0.f,0.f,0.f,0.f,0.f};
    v8f acc0 = zero, acc1 = zero, acc2 = zero, acc3 = zero;

    // prologue: stage 0 fragments
    v8i a_c  = load_a_frag(ap);
    v8i b0_c = load_b_frag(bp0);
    v8i b1_c = load_b_frag(bp1);
    v8i b2_c = load_b_frag(bp2);
    v8i b3_c = load_b_frag(bp3);

    int steps = K >> 5;     // 32-wide K steps; each step advances 16 dwords
    for (int s = 1; s < steps; ++s) {
        int d = s << 4;
        // issue next-step loads before consuming current fragments
        v8i a_n  = load_a_frag(ap  + d);
        v8i b0_n = load_b_frag(bp0 + d);
        v8i b1_n = load_b_frag(bp1 + d);
        v8i b2_n = load_b_frag(bp2 + d);
        v8i b3_n = load_b_frag(bp3 + d);

        acc0 = wmma_bf16(a_c, b0_c, acc0);
        acc1 = wmma_bf16(a_c, b1_c, acc1);
        acc2 = wmma_bf16(a_c, b2_c, acc2);
        acc3 = wmma_bf16(a_c, b3_c, acc3);

        a_c = a_n; b0_c = b0_n; b1_c = b1_n; b2_c = b2_n; b3_c = b3_n;
    }
    acc0 = wmma_bf16(a_c, b0_c, acc0);
    acc1 = wmma_bf16(a_c, b1_c, acc1);
    acc2 = wmma_bf16(a_c, b2_c, acc2);
    acc3 = wmma_bf16(a_c, b3_c, acc3);

    int cc = lane & 15, rb = (lane >> 4) << 3;
    float* Cp = C + (long long)(m0 + rb) * Ncol + n0 + cc;
#pragma unroll
    for (int v = 0; v < 8; ++v) {
        float* row = Cp + (long long)v * Ncol;
        row[0]  = acc0[v];
        row[16] = acc1[v];
        row[32] = acc2[v];
        row[48] = acc3[v];
    }
}

// ---------- attention ----------
// wave per (node, head): s_src/s_dst dot reductions
__global__ void k_attn_scores(const float* __restrict__ Hb, const float* __restrict__ a_src,
                              const float* __restrict__ a_dst, float* __restrict__ sS,
                              float* __restrict__ sD, int n, int H, int O, int ld) {
    int wid  = (int)((blockIdx.x * (long long)blockDim.x + threadIdx.x) >> 5);
    int lane = threadIdx.x & 31;
    if (wid >= n * H) return;
    int node = wid / H, h = wid % H;
    const float* hp = Hb + (long long)node * ld + h * O;
    float ss = 0.f, sd = 0.f;
    for (int o = lane; o < O; o += 32) {
        float hv = hp[o];
        ss += hv * a_src[h * O + o];
        sd += hv * a_dst[h * O + o];
    }
    for (int m = 16; m; m >>= 1) { ss += __shfl_xor(ss, m, 32); sd += __shfl_xor(sd, m, 32); }
    if (lane == 0) { sS[wid] = ss; sD[wid] = sd; }
}

__global__ void k_fill_u32(unsigned* p, unsigned v, long long n) {
    for (long long i = blockIdx.x * (long long)blockDim.x + threadIdx.x; i < n;
         i += (long long)gridDim.x * blockDim.x) p[i] = v;
}

__device__ __forceinline__ void edge_sd(const int* ei, long long e, int& s, int& d) {
    if (e < EE) { s = ei[e]; d = ei[EE + e]; }
    else        { s = (int)(e - EE); d = s; }
}

__global__ void k_edge_max(const int* __restrict__ ei, const float* __restrict__ sS,
                           const float* __restrict__ sD, unsigned* __restrict__ m, int H) {
    long long total = (long long)EA * H;
    for (long long i = blockIdx.x * (long long)blockDim.x + threadIdx.x; i < total;
         i += (long long)gridDim.x * blockDim.x) {
        long long e = i / H; int h = (int)(i % H);
        int s, d; edge_sd(ei, e, s, d);
        float v = sS[(long long)s * H + h] + sD[(long long)d * H + h];
        v = v > 0.f ? v : 0.2f * v;                        // leaky relu
        atomicMax(&m[(long long)d * H + h], enc_f(v));
    }
}

__global__ void k_edge_exp(const int* __restrict__ ei, const float* __restrict__ sS,
                           const float* __restrict__ sD, const unsigned* __restrict__ m,
                           float* __restrict__ ex, float* __restrict__ den, int H) {
    long long total = (long long)EA * H;
    for (long long i = blockIdx.x * (long long)blockDim.x + threadIdx.x; i < total;
         i += (long long)gridDim.x * blockDim.x) {
        long long e = i / H; int h = (int)(i % H);
        int s, d; edge_sd(ei, e, s, d);
        float v = sS[(long long)s * H + h] + sD[(long long)d * H + h];
        v = v > 0.f ? v : 0.2f * v;
        float w = __expf(v - dec_f(m[(long long)d * H + h]));
        ex[i] = w;
        atomicAdd(&den[(long long)d * H + h], w);
    }
}

__global__ void k_edge_norm(const int* __restrict__ ei, const float* __restrict__ den,
                            float* __restrict__ ex, int H) {
    long long total = (long long)EA * H;
    for (long long i = blockIdx.x * (long long)blockDim.x + threadIdx.x; i < total;
         i += (long long)gridDim.x * blockDim.x) {
        long long e = i / H; int h = (int)(i % H);
        int s, d; edge_sd(ei, e, s, d);
        ex[i] = ex[i] / den[(long long)d * H + h];
    }
}

__global__ void k_init_out(float* __restrict__ out, const float* __restrict__ bias,
                           int n, int D, int ld) {
    long long total = (long long)n * ld;
    for (long long i = blockIdx.x * (long long)blockDim.x + threadIdx.x; i < total;
         i += (long long)gridDim.x * blockDim.x) {
        int c = (int)(i % ld);
        out[i] = (c < D) ? bias[c] : 0.f;
    }
}

__global__ void k_aggregate(const int* __restrict__ ei, const float* __restrict__ ex,
                            const float* __restrict__ Hb, float* __restrict__ out,
                            int H, int O, int ld) {
    long long total = (long long)EA * H * O;
    for (long long i = blockIdx.x * (long long)blockDim.x + threadIdx.x; i < total;
         i += (long long)gridDim.x * blockDim.x) {
        int o = (int)(i % O);
        long long t = i / O;
        int h = (int)(t % H);
        long long e = t / H;
        int s, d; edge_sd(ei, e, s, d);
        float a = ex[e * H + h];
        atomicAdd(&out[(long long)d * ld + h * O + o],
                  a * Hb[(long long)s * ld + h * O + o]);
    }
}

// ---------- layernorm + relu (block per node, in-place) ----------
__global__ void k_ln_relu(float* __restrict__ x, const float* __restrict__ g,
                          const float* __restrict__ b, int D, int ld) {
    __shared__ float red[256];
    float* xp = x + (long long)blockIdx.x * ld;
    float s = 0.f, ss = 0.f;
    for (int c = threadIdx.x; c < D; c += blockDim.x) { float v = xp[c]; s += v; ss += v * v; }
    red[threadIdx.x] = s; __syncthreads();
    for (int st = 128; st; st >>= 1) { if ((int)threadIdx.x < st) red[threadIdx.x] += red[threadIdx.x + st]; __syncthreads(); }
    float mean = red[0] / D; __syncthreads();
    red[threadIdx.x] = ss; __syncthreads();
    for (int st = 128; st; st >>= 1) { if ((int)threadIdx.x < st) red[threadIdx.x] += red[threadIdx.x + st]; __syncthreads(); }
    float var = red[0] / D - mean * mean;
    float inv = rsqrtf(var + 1e-5f);
    __syncthreads();
    for (int c = threadIdx.x; c < D; c += blockDim.x) {
        float v = (xp[c] - mean) * inv * g[c] + b[c];
        xp[c] = v > 0.f ? v : 0.f;
    }
}

__global__ void k_cast_bf16(const float* __restrict__ x, unsigned short* __restrict__ y,
                            long long total) {
    for (long long i = blockIdx.x * (long long)blockDim.x + threadIdx.x; i < total;
         i += (long long)gridDim.x * blockDim.x) y[i] = f2bf(x[i]);
}

// ---------- pooling ----------
__global__ void k_vec_invnorm(const float* __restrict__ p, float* __restrict__ invn, int D) {
    __shared__ float red[256];
    float s = 0.f;
    for (int c = threadIdx.x; c < D; c += blockDim.x) s += p[c] * p[c];
    red[threadIdx.x] = s; __syncthreads();
    for (int st = 128; st; st >>= 1) { if ((int)threadIdx.x < st) red[threadIdx.x] += red[threadIdx.x + st]; __syncthreads(); }
    if (threadIdx.x == 0) invn[0] = rsqrtf(red[0]);
}

// wave per node: score[n] = (x[n,:].p) * invnorm
__global__ void k_pool_score(const float* __restrict__ x, const float* __restrict__ p,
                             const float* __restrict__ invn, float* __restrict__ score,
                             int n, int D, int ld) {
    int wid  = (int)((blockIdx.x * (long long)blockDim.x + threadIdx.x) >> 5);
    int lane = threadIdx.x & 31;
    if (wid >= n) return;
    const float* xp = x + (long long)wid * ld;
    float s = 0.f;
    for (int c = lane; c < D; c += 32) s += xp[c] * p[c];
    for (int m = 16; m; m >>= 1) s += __shfl_xor(s, m, 32);
    if (lane == 0) score[wid] = s * invn[0];
}

// single-workgroup bitonic sort (descending) over SORTN padded scores in LDS
__global__ void k_bitonic_topk(const float* __restrict__ score, int n, int SORTN,
                               float* __restrict__ vals_out, int* __restrict__ idx_out) {
    extern __shared__ char smem[];
    float* sv = (float*)smem;
    int*   si = (int*)(smem + (size_t)SORTN * 4);
    for (int i = threadIdx.x; i < SORTN; i += blockDim.x) {
        sv[i] = (i < n) ? score[i] : -3.4e38f;
        si[i] = i;
    }
    __syncthreads();
    for (int k = 2; k <= SORTN; k <<= 1) {
        for (int j = k >> 1; j > 0; j >>= 1) {
            for (int i = threadIdx.x; i < SORTN; i += blockDim.x) {
                int ixj = i ^ j;
                if (ixj > i) {
                    bool desc = ((i & k) == 0);
                    float a = sv[i], c = sv[ixj];
                    bool sw = desc ? (a < c) : (a > c);
                    if (sw) {
                        sv[i] = c; sv[ixj] = a;
                        int t = si[i]; si[i] = si[ixj]; si[ixj] = t;
                    }
                }
            }
            __syncthreads();
        }
    }
    for (int i = threadIdx.x; i < SORTN; i += blockDim.x) { vals_out[i] = sv[i]; idx_out[i] = si[i]; }
}

__global__ void k_gather(const float* __restrict__ xin, const float* __restrict__ vals,
                         const int* __restrict__ idx, float* __restrict__ xout,
                         int k, int D, int ld) {
    long long total = (long long)k * D;
    for (long long i = blockIdx.x * (long long)blockDim.x + threadIdx.x; i < total;
         i += (long long)gridDim.x * blockDim.x) {
        int r = (int)(i / D), c = (int)(i % D);
        xout[(long long)r * ld + c] = xin[(long long)idx[r] * ld + c] * tanhf(vals[r]);
    }
}

__global__ void k_mean(const float* __restrict__ x, float* __restrict__ pooled,
                       int rows, int D, int ld) {
    for (int c = blockIdx.x * blockDim.x + threadIdx.x; c < D; c += gridDim.x * blockDim.x) {
        float s = 0.f;
        for (int r = 0; r < rows; ++r) s += x[(long long)r * ld + c];
        pooled[c] = s / rows;
    }
}

__global__ void k_final(const float* __restrict__ pooled, const float* __restrict__ Wout,
                        const float* __restrict__ bout, float* __restrict__ out,
                        int D, int Dout) {
    for (int j = blockIdx.x * blockDim.x + threadIdx.x; j < Dout; j += gridDim.x * blockDim.x) {
        float s = bout[j];
        for (int c = 0; c < D; ++c) s += pooled[c] * Wout[(long long)c * Dout + j];
        out[j] = s;
    }
}

// ---------- host launch ----------
extern "C" void kernel_launch(void* const* d_in, const int* in_sizes, int n_in,
                              void* d_out, int out_size, void* d_ws, size_t ws_size,
                              hipStream_t stream) {
    (void)in_sizes; (void)n_in; (void)out_size; (void)ws_size;
    const float* tok    = (const float*)d_in[0];
    const float* pos    = (const float*)d_in[1];
    const int*   ei     = (const int*)  d_in[2];
    // d_in[3] edge_weight unused (GATConv has no edge features)
    const float* W1     = (const float*)d_in[4];
    const float* a1s    = (const float*)d_in[5];
    const float* a1d    = (const float*)d_in[6];
    const float* b1     = (const float*)d_in[7];
    const float* ln1g   = (const float*)d_in[8];
    const float* ln1b   = (const float*)d_in[9];
    const float* W2     = (const float*)d_in[10];
    const float* a2s    = (const float*)d_in[11];
    const float* a2d    = (const float*)d_in[12];
    const float* b2     = (const float*)d_in[13];
    const float* ln2g   = (const float*)d_in[14];
    const float* ln2b   = (const float*)d_in[15];
    const float* p1     = (const float*)d_in[16];
    const float* p2     = (const float*)d_in[17];
    const float* p3     = (const float*)d_in[18];
    const float* Wout   = (const float*)d_in[19];
    const float* bout   = (const float*)d_in[20];
    float* out = (float*)d_out;

    // workspace carve (deterministic)
    char* w = (char*)d_ws;
    size_t off = 0;
    auto take = [&](size_t bytes) -> void* {
        void* p = w + off;
        off = (off + bytes + 255) & ~(size_t)255;
        return p;
    };
    unsigned short* X1  = (unsigned short*)take((size_t)NN * K1P * 2);
    unsigned short* W1t = (unsigned short*)take((size_t)DMID * K1P * 2);
    unsigned short* W2t = (unsigned short*)take((size_t)LD2 * K2K * 2);
    float* RH = (float*)take((size_t)NN * DMID * 4);   // H1 -> X2(bf16) -> pool ping
    float* RO = (float*)take((size_t)NN * DMID * 4);   // out1/LN1 -> H2
    float* RP = (float*)take((size_t)NN * LD2 * 4);    // out2/LN2 -> pool pong
    float*    sS  = (float*)   take((size_t)NN * H1H * 4);
    float*    sD  = (float*)   take((size_t)NN * H1H * 4);
    unsigned* mB  = (unsigned*)take((size_t)NN * H1H * 4);
    float*    den = (float*)   take((size_t)NN * H1H * 4);
    float*    ex  = (float*)   take((size_t)EA * H1H * 4);
    float* score  = (float*)take(16384 * 4);
    float* svals  = (float*)take(16384 * 4);
    int*   sidx   = (int*)  take(16384 * 4);
    float* invn   = (float*)take(256);
    float* pooled = (float*)take((size_t)LD2 * 4);

    const int B = 256;
    const int GS = 4096;   // generic grid-stride grid

    // ---- stage 0: pack inputs / weights to bf16 ----
    k_build_x1<<<GS, B, 0, stream>>>(tok, pos, X1);
    k_prep_wt <<<GS, B, 0, stream>>>(W1, W1t, DMID, 481, DMID, K1P);
    k_prep_wt <<<GS, B, 0, stream>>>(W2, W2t, DPOOL, K2K, LD2, K2K);

    // ---- GEMM1: H1[N,2048] = X1 @ W1 ----
    {
        int waves = (NN / 16) * (DMID / 64);
        int blocks = (waves * 32 + B - 1) / B;
        k_gemm_bf16<<<blocks, B, 0, stream>>>(X1, W1t, RH, NN, DMID, K1P);
    }

    // ---- GAT layer 1 ----
    {
        int waves = NN * H1H;
        k_attn_scores<<<(waves * 32 + B - 1) / B, B, 0, stream>>>(RH, a1s, a1d, sS, sD, NN, H1H, O1O, DMID);
        k_fill_u32<<<GS, B, 0, stream>>>(mB, 0u, (long long)NN * H1H);
        k_fill_u32<<<GS, B, 0, stream>>>((unsigned*)den, 0u, (long long)NN * H1H);
        k_edge_max <<<GS, B, 0, stream>>>(ei, sS, sD, mB, H1H);
        k_edge_exp <<<GS, B, 0, stream>>>(ei, sS, sD, mB, ex, den, H1H);
        k_edge_norm<<<GS, B, 0, stream>>>(ei, den, ex, H1H);
        k_init_out <<<GS, B, 0, stream>>>(RO, b1, NN, DMID, DMID);
        k_aggregate<<<8192, B, 0, stream>>>(ei, ex, RH, RO, H1H, O1O, DMID);
        k_ln_relu  <<<NN, B, 0, stream>>>(RO, ln1g, ln1b, DMID, DMID);
    }

    // ---- cast x -> bf16 (into RH; H1 is dead), GEMM2: H2[N,1472] = X2 @ W2 ----
    k_cast_bf16<<<GS, B, 0, stream>>>(RO, (unsigned short*)RH, (long long)NN * DMID);
    {
        int waves = (NN / 16) * (LD2 / 64);
        int blocks = (waves * 32 + B - 1) / B;
        k_gemm_bf16<<<blocks, B, 0, stream>>>((unsigned short*)RH, W2t, RO, NN, LD2, K2K);
    }

    // ---- GAT layer 2 ----
    {
        int waves = NN * H2H;
        k_attn_scores<<<(waves * 32 + B - 1) / B, B, 0, stream>>>(RO, a2s, a2d, sS, sD, NN, H2H, O2O, LD2);
        k_fill_u32<<<GS, B, 0, stream>>>(mB, 0u, (long long)NN * H2H);
        k_fill_u32<<<GS, B, 0, stream>>>((unsigned*)den, 0u, (long long)NN * H2H);
        k_edge_max <<<GS, B, 0, stream>>>(ei, sS, sD, mB, H2H);
        k_edge_exp <<<GS, B, 0, stream>>>(ei, sS, sD, mB, ex, den, H2H);
        k_edge_norm<<<GS, B, 0, stream>>>(ei, den, ex, H2H);
        k_init_out <<<GS, B, 0, stream>>>(RP, b2, NN, DPOOL, LD2);
        k_aggregate<<<8192, B, 0, stream>>>(ei, ex, RO, RP, H2H, O2O, LD2);
        k_ln_relu  <<<NN, B, 0, stream>>>(RP, ln2g, ln2b, DPOOL, LD2);
    }

    // ---- hierarchical TopK pooling: RP(10000) -> RH(5000) -> RP(1500) -> RH(300) ----
    struct PoolStep { const float* p; int nin; int k; int sortn; };
    const PoolStep steps[3] = { {p1, NN, 5000, 16384}, {p2, 5000, 1500, 8192}, {p3, 1500, 300, 2048} };
    float* bufs[2] = { RP, RH };
    for (int sidx2 = 0; sidx2 < 3; ++sidx2) {
        const PoolStep& st = steps[sidx2];
        float* xin  = bufs[sidx2 & 1];
        float* xout = bufs[(sidx2 + 1) & 1];
        k_vec_invnorm<<<1, B, 0, stream>>>(st.p, invn, DPOOL);
        k_pool_score <<<(st.nin * 32 + B - 1) / B, B, 0, stream>>>(xin, st.p, invn, score, st.nin, DPOOL, LD2);
        k_bitonic_topk<<<1, 1024, (size_t)st.sortn * 8, stream>>>(score, st.nin, st.sortn, svals, sidx);
        k_gather<<<GS, B, 0, stream>>>(xin, svals, sidx, xout, st.k, DPOOL, LD2);
    }

    // ---- global mean pool + output projection ----
    k_mean <<<8, B, 0, stream>>>(RH, pooled, 300, DPOOL, LD2);
    k_final<<<2, B, 0, stream>>>(pooled, Wout, bout, out, DPOOL, 481);
}
